// _RuntimeLongcatRouter_78752520339554
// MI455X (gfx1250) — matrix operand order repaired and checked by reading, not compiled, not run on hardware
//
#include <hip/hip_runtime.h>
#include <hip/hip_bf16.h>

// GEMM: out[t,e] = sum_h X[t,h] * W[e,h],  X:[8192,6144] bf16, W:[768,6144] bf16, out fp32.
// Compute-bound (AI ~573 FLOP/B) -> v_wmma_f32_16x16x32_bf16 dense matrix pipe.

typedef __attribute__((ext_vector_type(16))) __bf16 v16bf;
typedef __attribute__((ext_vector_type(8)))  __bf16 v8bf;
typedef __attribute__((ext_vector_type(8)))  float  v8f;

#define TOKENS 8192
#define HIDDEN 6144
#define LOGITS 768

#define BLOCK_M 128
#define BLOCK_N 128
#define WAVE_M  64
#define WAVE_N  32
#define MT (WAVE_M / 16)   // 4 m-tiles per wave
#define NT (WAVE_N / 16)   // 2 n-tiles per wave
#define KSTEP 32           // WMMA K depth for bf16

union Frag {
    struct { v8bf lo, hi; } p;   // two 16B contiguous chunks per lane
    v16bf v;
};

__global__ __launch_bounds__(256)
void router_logits_wmma(const __bf16* __restrict__ X,
                        const __bf16* __restrict__ W,
                        float* __restrict__ out)
{
    const int lane  = threadIdx.x & 31;
    const int wave  = threadIdx.x >> 5;
    const int waveM = wave >> 2;        // 0..1
    const int waveN = wave & 3;         // 0..3
    const int l16   = lane & 15;
    const int half  = lane >> 4;        // 0 or 1 (lane half)

    const int rowBase = blockIdx.x * BLOCK_M + waveM * WAVE_M;
    const int colBase = blockIdx.y * BLOCK_N + waveN * WAVE_N;

    // A fragment (16x32 bf16, ISA layout): lane half 0 -> K in [0,8)+[16,24),
    // lane half 1 -> K in [8,16)+[24,32). Row = l16 within the 16x16 m-tile.
    const __bf16* aptr[MT];
#pragma unroll
    for (int mt = 0; mt < MT; ++mt)
        aptr[mt] = X + (size_t)(rowBase + mt * 16 + l16) * HIDDEN + half * 8;

    // B fragment (32x16 bf16, ISA layout): lane half 0 -> K in [0,8)+[16,24),
    // lane half 1 -> K in [8,16)+[24,32). Column (N) = l16. W rows are K-contiguous.
    const __bf16* bptr[NT];
#pragma unroll
    for (int nt = 0; nt < NT; ++nt)
        bptr[nt] = W + (size_t)(colBase + nt * 16 + l16) * HIDDEN + half * 8;

    v8f acc[MT][NT];
    const v8f vzero = {0.f, 0.f, 0.f, 0.f, 0.f, 0.f, 0.f, 0.f};
#pragma unroll
    for (int mt = 0; mt < MT; ++mt)
#pragma unroll
        for (int nt = 0; nt < NT; ++nt)
            acc[mt][nt] = vzero;

    for (int k = 0; k < HIDDEN; k += KSTEP) {
        Frag a[MT], b[NT];
#pragma unroll
        for (int mt = 0; mt < MT; ++mt) {
            a[mt].p.lo = *(const v8bf*)(aptr[mt] + k);        // 16B aligned
            a[mt].p.hi = *(const v8bf*)(aptr[mt] + k + 16);   // 16B aligned
            // Prefetch the streamed operand (X misses L2; W stays resident).
            __builtin_prefetch(aptr[mt] + k + 4 * KSTEP, 0, 0);
        }
#pragma unroll
        for (int nt = 0; nt < NT; ++nt) {
            b[nt].p.lo = *(const v8bf*)(bptr[nt] + k);        // 16B aligned
            b[nt].p.hi = *(const v8bf*)(bptr[nt] + k + 16);   // 16B aligned
        }
#pragma unroll
        for (int mt = 0; mt < MT; ++mt)
#pragma unroll
            for (int nt = 0; nt < NT; ++nt)
                acc[mt][nt] = __builtin_amdgcn_wmma_f32_16x16x32_bf16(
                    /*neg_a=*/false, a[mt].v,
                    /*neg_b=*/false, b[nt].v,
                    /*c_mod=*/(short)0, acc[mt][nt],
                    /*reuse_a=*/false, /*reuse_b=*/false);
    }

    // C/D layout (16x16 f32): VGPR j -> row = j + 8*half, col = l16.
#pragma unroll
    for (int mt = 0; mt < MT; ++mt) {
#pragma unroll
        for (int nt = 0; nt < NT; ++nt) {
            const int row0 = rowBase + mt * 16 + half * 8;
            const int col  = colBase + nt * 16 + l16;
#pragma unroll
            for (int j = 0; j < 8; ++j)
                out[(size_t)(row0 + j) * LOGITS + col] = acc[mt][nt][j];
        }
    }
}

extern "C" void kernel_launch(void* const* d_in, const int* in_sizes, int n_in,
                              void* d_out, int out_size, void* d_ws, size_t ws_size,
                              hipStream_t stream) {
    const __bf16* X = (const __bf16*)d_in[0];   // hidden_states [8192, 6144] bf16
    const __bf16* W = (const __bf16*)d_in[1];   // weight        [768, 6144] bf16
    float* out = (float*)d_out;                 // [8192, 768] fp32

    dim3 grid(TOKENS / BLOCK_M, LOGITS / BLOCK_N);   // 64 x 6 = 384 workgroups
    router_logits_wmma<<<grid, 256, 0, stream>>>(X, W, out);
}
